// RNN_LSTM2_truncated_90907277787358
// MI455X (gfx1250) — matrix operand order repaired and checked by reading, not compiled, not run on hardware
//
#include <hip/hip_runtime.h>
#include <hip/hip_bf16.h>

// ---------------------------------------------------------------------------
// 2-layer LSTM (B=256, T=784, H=512) + FC head, persistent-kernel design for
// MI455X (gfx1250, wave32, WMMA).
//
// Roofline: 1.26 TFLOP of bf16 GEMM; weights (6MB bf16) fit in L2 and each
// workgroup's slice (195KB) fits in the 320KB WGP LDS -> load weights to LDS
// once, keep cell state in fp32 VGPRs, only h/c state (256KB bf16) crosses
// workgroups per step via L2. Two grid barriers per timestep.
// ---------------------------------------------------------------------------

#define B_      256
#define T_      784
#define H_      512
#define C_      10
#define NWG     32
#define NSLICE  16              // H_ / NWG hidden columns per workgroup
#define THREADS 256             // 8 wave32 waves
#define LSTR    520             // padded LDS row stride (bf16 elems) -> conflict-free
#define BH      (B_ * H_)       // elements per state-buffer parity

typedef __attribute__((ext_vector_type(16))) __bf16 v16bf;
typedef __attribute__((ext_vector_type(8)))  float  v8f;

// ---- WMMA operand loaders -------------------------------------------------
// A operand (16x32 bf16, M on lanes): lane half h -> elems 0..7 = K k0+8h..,
// elems 8..15 = K k0+16+8h.. (two 16B chunks)
__device__ __forceinline__ v16bf load_a16(const __bf16* base, int row,
                                          int rowStride, int k0, int half) {
  const __bf16* p = base + (size_t)row * rowStride + k0 + 8 * half;
  v16bf a;
  ((uint4*)&a)[0] = *(const uint4*)(p);
  ((uint4*)&a)[1] = *(const uint4*)(p + 16);
  return a;
}

// B operand (32x16 bf16, N on lanes): lane half h -> elems 0..7 = K k0+8h..,
// elems 8..15 = K k0+16+8h.. (two 16B chunks)
__device__ __forceinline__ v16bf load_b16(const __bf16* base, int row,
                                          int rowStride, int k0, int half) {
  const __bf16* p = base + (size_t)row * rowStride + k0 + 8 * half;
  v16bf b;
  ((uint4*)&b)[0] = *(const uint4*)(p);
  ((uint4*)&b)[1] = *(const uint4*)(p + 16);
  return b;
}

__device__ __forceinline__ float sigf(float x) {
  return 1.0f / (1.0f + __expf(-x));
}

// ---- device-scope grid barrier (sense via generation counter) -------------
__device__ __forceinline__ void grid_barrier(int* sync, int target) {
  __threadfence();
  __syncthreads();
  if (threadIdx.x == 0) {
    int* cnt = sync;
    int* gen = sync + 64;  // separate cacheline
    int prev = __hip_atomic_fetch_add(cnt, 1, __ATOMIC_ACQ_REL,
                                      __HIP_MEMORY_SCOPE_AGENT);
    if (prev == NWG - 1) {
      __hip_atomic_store(cnt, 0, __ATOMIC_RELAXED, __HIP_MEMORY_SCOPE_AGENT);
      __hip_atomic_store(gen, target, __ATOMIC_RELEASE,
                         __HIP_MEMORY_SCOPE_AGENT);
    } else {
      while (__hip_atomic_load(gen, __ATOMIC_ACQUIRE,
                               __HIP_MEMORY_SCOPE_AGENT) < target) {
        __builtin_amdgcn_s_sleep(1);
      }
    }
  }
  __syncthreads();
  __threadfence();
}

// ---- FC head: out[:, t, :] = h2(t) @ W_fc^T + b_fc (one 16-row M tile) ----
__device__ __forceinline__ void fc_head(const __bf16* h2src, const __bf16* Wfcb,
                                        float* out, int t, int mb, int half,
                                        int n, float bfc_n) {
  v8f acc;
#pragma unroll
  for (int v = 0; v < 8; ++v) acc[v] = bfc_n;
#pragma unroll 4
  for (int k0 = 0; k0 < H_; k0 += 32) {
    v16bf a = load_a16(h2src, mb + n, H_, k0, half);
    v16bf b = load_b16(Wfcb, n, H_, k0, half);
    acc = __builtin_amdgcn_wmma_f32_16x16x32_bf16(false, a, false, b, (short)0,
                                                  acc, false, false);
  }
  if (n < C_) {
#pragma unroll
    for (int v = 0; v < 8; ++v) {
      int m = mb + v + 8 * half;
      out[(size_t)m * (T_ * C_) + t * C_ + n] = acc[v];
    }
  }
}

// ---------------------------------------------------------------------------
// Prep: fp32 -> bf16 weight conversion, state zero-init, barrier reset.
// Re-run every launch so graph replays are deterministic.
// ---------------------------------------------------------------------------
__global__ void lstm2_prep(const float* __restrict__ Whh1,
                           const float* __restrict__ Wih2,
                           const float* __restrict__ Whh2,
                           const float* __restrict__ Wfc,
                           __bf16* __restrict__ Whh1b,
                           __bf16* __restrict__ Wih2b,
                           __bf16* __restrict__ Whh2b,
                           __bf16* __restrict__ Wfcb,
                           __bf16* __restrict__ h1buf,
                           __bf16* __restrict__ c1buf,
                           __bf16* __restrict__ h2buf, int* __restrict__ sync) {
  const int stride = gridDim.x * blockDim.x;
  const int tid0 = blockIdx.x * blockDim.x + threadIdx.x;
  for (int i = tid0; i < 4 * H_ * H_; i += stride) {
    Whh1b[i] = (__bf16)Whh1[i];
    Wih2b[i] = (__bf16)Wih2[i];
    Whh2b[i] = (__bf16)Whh2[i];
  }
  for (int i = tid0; i < 16 * H_; i += stride) {
    int r = i >> 9;
    Wfcb[i] = (r < C_) ? (__bf16)Wfc[i] : (__bf16)0.0f;  // zero-pad rows 10..15
  }
  for (int i = tid0; i < 2 * BH; i += stride) {
    h1buf[i] = (__bf16)0.0f;
    c1buf[i] = (__bf16)0.0f;
    h2buf[i] = (__bf16)0.0f;
  }
  if (tid0 == 0) { sync[0] = 0; sync[64] = 0; }
}

// ---------------------------------------------------------------------------
// Persistent LSTM kernel: 32 WGs x 8 waves, weights resident in LDS.
// ---------------------------------------------------------------------------
__global__ void __launch_bounds__(THREADS, 1)
lstm2_persistent(const float* __restrict__ x, const float* __restrict__ Wih1,
                 const float* __restrict__ bih1, const float* __restrict__ bhh1,
                 const float* __restrict__ bih2, const float* __restrict__ bhh2,
                 const float* __restrict__ bfc,
                 const __bf16* __restrict__ Whh1b,
                 const __bf16* __restrict__ Wih2b,
                 const __bf16* __restrict__ Whh2b,
                 const __bf16* __restrict__ Wfcb, __bf16* __restrict__ h1buf,
                 __bf16* __restrict__ c1buf, __bf16* __restrict__ h2buf,
                 float* __restrict__ out, int* __restrict__ sync) {
  extern __shared__ __bf16 smem[];
  __bf16* Wl1  = smem;               // W_hh1 slice: 64 rows (4 gates x 16) x 512
  __bf16* Wl2i = smem + 64 * LSTR;   // W_ih2 slice
  __bf16* Wl2h = smem + 128 * LSTR;  // W_hh2 slice

  const int wg = blockIdx.x;
  const int tid = threadIdx.x;
  const int wave = tid >> 5;
  const int lane = tid & 31;
  const int half = lane >> 4;
  const int n = lane & 15;                 // N index within a 16-wide tile
  const int mbase = wave * 32;             // this wave's 32 batch rows
  const int ncol = wg * NSLICE + n;        // hidden column owned by this lane

  // ---- one-time LDS preload of this WG's weight slices --------------------
  {
    const __bf16* srcs[3] = {Whh1b, Wih2b, Whh2b};
#pragma unroll
    for (int a = 0; a < 3; ++a) {
      const __bf16* src = srcs[a];
      __bf16* dst = smem + a * 64 * LSTR;
      for (int i = tid * 8; i < 64 * 512; i += THREADS * 8) {
        int r = i >> 9;                    // LDS row 0..63
        int c = i & 511;
        int gate = r >> 4;
        int rr = r & 15;
        int srow = gate * H_ + wg * NSLICE + rr;
        *(uint4*)(dst + r * LSTR + c) = *(const uint4*)(src + (size_t)srow * H_ + c);
      }
    }
  }

  // per-lane constants: input-projection weight + fused biases per gate
  float wih1g[4], bias1[4], bias2[4];
#pragma unroll
  for (int g = 0; g < 4; ++g) {
    int row = g * H_ + ncol;
    wih1g[g] = Wih1[row];
    bias1[g] = bih1[row] + bhh1[row];
    bias2[g] = bih2[row] + bhh2[row];
  }
  float bfc_n = (n < C_) ? bfc[n] : 0.0f;

  __syncthreads();

  // cell state lives in fp32 VGPRs for all 784 steps
  v8f c1r[2] = {};
  v8f c2r[2] = {};

  int rp = 0;   // read parity of state buffers
  int gen = 0;  // grid-barrier generation

  for (int t = 0; t < T_; ++t) {
    const int wp = rp ^ 1;

    // ---- overlapped FC head for step t-1 (h2(t-1) is in buf[rp]) ----------
    if (t > 0 && wg < 16 && wave == 0)
      fc_head(h2buf + rp * BH, Wfcb, out, t - 1, wg * 16, half, n, bfc_n);

    // ======================= layer 1: gates = pre1 + h1 W_hh1^T ============
    v8f acc[2][4];
#pragma unroll
    for (int j = 0; j < 2; ++j)
#pragma unroll
      for (int g = 0; g < 4; ++g)
#pragma unroll
        for (int v = 0; v < 8; ++v) {
          int m = mbase + j * 16 + v + 8 * half;
          acc[j][g][v] = x[(size_t)m * T_ + t] * wih1g[g] + bias1[g];
        }

    {
      const __bf16* h1r = h1buf + rp * BH;
#pragma unroll 4
      for (int k0 = 0; k0 < H_; k0 += 32) {
        v16bf a0 = load_a16(h1r, mbase + n, H_, k0, half);
        v16bf a1 = load_a16(h1r, mbase + 16 + n, H_, k0, half);
#pragma unroll
        for (int g = 0; g < 4; ++g) {
          v16bf b = load_b16(Wl1, g * 16 + n, LSTR, k0, half);
          acc[0][g] = __builtin_amdgcn_wmma_f32_16x16x32_bf16(
              false, a0, false, b, (short)0, acc[0][g], false, false);
          acc[1][g] = __builtin_amdgcn_wmma_f32_16x16x32_bf16(
              false, a1, false, b, (short)0, acc[1][g], false, false);
        }
      }
    }

    // ---- cell 1 elementwise; publish h1,c1 as bf16 ------------------------
    {
      __bf16* h1w = h1buf + wp * BH;
      __bf16* c1w = c1buf + wp * BH;
#pragma unroll
      for (int j = 0; j < 2; ++j)
#pragma unroll
        for (int v = 0; v < 8; ++v) {
          float iv = sigf(acc[j][0][v]);
          float fv = sigf(acc[j][1][v]);
          float gv = tanhf(acc[j][2][v]);
          float ov = sigf(acc[j][3][v]);
          float c = fv * c1r[j][v] + iv * gv;
          c1r[j][v] = c;
          float h = ov * tanhf(c);
          int m = mbase + j * 16 + v + 8 * half;
          h1w[(size_t)m * H_ + ncol] = (__bf16)h;
          c1w[(size_t)m * H_ + ncol] = (__bf16)c;
        }
    }

    grid_barrier(sync, ++gen);  // c1(t), h1(t) visible grid-wide

    // ============ layer 2: gates = c1 W_ih2^T + h2 W_hh2^T + b =============
#pragma unroll
    for (int j = 0; j < 2; ++j)
#pragma unroll
      for (int g = 0; g < 4; ++g)
#pragma unroll
        for (int v = 0; v < 8; ++v) acc[j][g][v] = bias2[g];

    {
      const __bf16* c1rd = c1buf + wp * BH;
#pragma unroll 4
      for (int k0 = 0; k0 < H_; k0 += 32) {
        v16bf a0 = load_a16(c1rd, mbase + n, H_, k0, half);
        v16bf a1 = load_a16(c1rd, mbase + 16 + n, H_, k0, half);
#pragma unroll
        for (int g = 0; g < 4; ++g) {
          v16bf b = load_b16(Wl2i, g * 16 + n, LSTR, k0, half);
          acc[0][g] = __builtin_amdgcn_wmma_f32_16x16x32_bf16(
              false, a0, false, b, (short)0, acc[0][g], false, false);
          acc[1][g] = __builtin_amdgcn_wmma_f32_16x16x32_bf16(
              false, a1, false, b, (short)0, acc[1][g], false, false);
        }
      }
    }
    {
      const __bf16* h2rd = h2buf + rp * BH;
#pragma unroll 4
      for (int k0 = 0; k0 < H_; k0 += 32) {
        v16bf a0 = load_a16(h2rd, mbase + n, H_, k0, half);
        v16bf a1 = load_a16(h2rd, mbase + 16 + n, H_, k0, half);
#pragma unroll
        for (int g = 0; g < 4; ++g) {
          v16bf b = load_b16(Wl2h, g * 16 + n, LSTR, k0, half);
          acc[0][g] = __builtin_amdgcn_wmma_f32_16x16x32_bf16(
              false, a0, false, b, (short)0, acc[0][g], false, false);
          acc[1][g] = __builtin_amdgcn_wmma_f32_16x16x32_bf16(
              false, a1, false, b, (short)0, acc[1][g], false, false);
        }
      }
    }

    // ---- cell 2 elementwise; publish h2 -----------------------------------
    {
      __bf16* h2w = h2buf + wp * BH;
#pragma unroll
      for (int j = 0; j < 2; ++j)
#pragma unroll
        for (int v = 0; v < 8; ++v) {
          float iv = sigf(acc[j][0][v]);
          float fv = sigf(acc[j][1][v]);
          float gv = tanhf(acc[j][2][v]);
          float ov = sigf(acc[j][3][v]);
          float c = fv * c2r[j][v] + iv * gv;
          c2r[j][v] = c;
          float h = ov * tanhf(c);
          int m = mbase + j * 16 + v + 8 * half;
          h2w[(size_t)m * H_ + ncol] = (__bf16)h;
        }
    }

    grid_barrier(sync, ++gen);  // h2(t) visible grid-wide
    rp = wp;
  }

  // ---- FC head for the final step -----------------------------------------
  if (wg < 16 && wave == 0)
    fc_head(h2buf + rp * BH, Wfcb, out, T_ - 1, wg * 16, half, n, bfc_n);
}

// ---------------------------------------------------------------------------
extern "C" void kernel_launch(void* const* d_in, const int* in_sizes, int n_in,
                              void* d_out, int out_size, void* d_ws,
                              size_t ws_size, hipStream_t stream) {
  (void)in_sizes; (void)n_in; (void)out_size; (void)ws_size;

  const float* x    = (const float*)d_in[0];
  const float* Wih1 = (const float*)d_in[1];
  const float* Whh1 = (const float*)d_in[2];
  const float* bih1 = (const float*)d_in[3];
  const float* bhh1 = (const float*)d_in[4];
  const float* Wih2 = (const float*)d_in[5];
  const float* Whh2 = (const float*)d_in[6];
  const float* bih2 = (const float*)d_in[7];
  const float* bhh2 = (const float*)d_in[8];
  const float* Wfc  = (const float*)d_in[9];
  const float* bfc  = (const float*)d_in[10];
  float* out = (float*)d_out;

  // workspace carve-up (~7.9 MB total)
  char* ws = (char*)d_ws;
  size_t off = 0;
  auto carve = [&](size_t bytes) -> char* {
    char* p = ws + off;
    off += (bytes + 255) & ~(size_t)255;
    return p;
  };
  int*    syncp = (int*)carve(4096);
  __bf16* h1buf = (__bf16*)carve((size_t)2 * BH * sizeof(__bf16));
  __bf16* c1buf = (__bf16*)carve((size_t)2 * BH * sizeof(__bf16));
  __bf16* h2buf = (__bf16*)carve((size_t)2 * BH * sizeof(__bf16));
  __bf16* Whh1b = (__bf16*)carve((size_t)4 * H_ * H_ * sizeof(__bf16));
  __bf16* Wih2b = (__bf16*)carve((size_t)4 * H_ * H_ * sizeof(__bf16));
  __bf16* Whh2b = (__bf16*)carve((size_t)4 * H_ * H_ * sizeof(__bf16));
  __bf16* Wfcb  = (__bf16*)carve((size_t)16 * H_ * sizeof(__bf16));

  lstm2_prep<<<512, 256, 0, stream>>>(Whh1, Wih2, Whh2, Wfc, Whh1b, Wih2b,
                                      Whh2b, Wfcb, h1buf, c1buf, h2buf, syncp);

  const size_t smem = (size_t)192 * LSTR * sizeof(__bf16);  // 199,680 B < 320 KB
  lstm2_persistent<<<NWG, THREADS, smem, stream>>>(
      x, Wih1, bih1, bhh1, bih2, bhh2, bfc, Whh1b, Wih2b, Whh2b, Wfcb, h1buf,
      c1buf, h2buf, out, syncp);
}
